// raw_BiLSTM_27384711479309
// MI455X (gfx1250) — compile-verified, matrix operand order checked by reading, not run on hardware
//
#include <hip/hip_runtime.h>
#include <hip/hip_bf16.h>
#include <stdint.h>

// BiLSTM (V=256,H=1024,L=2,DEPTH=2,B=64,S=128) for MI455X gfx1250.
// Strategy: bf16 weights (~113MB -> resident in 192MB L2, so per-timestep
// weight re-reads hit L2 not HBM), v_wmma_f32_16x16x32_bf16 GEMMs with
// full-batch (M=64) register reuse of each weight tile (4 WMMAs per B load),
// explicit 2-stage ping-pong k-loop (no rotation moves, loads outstanding
// across WMMAs), 8-wave blocks with K-split wave pairs (latency hiding at
// zero extra L2 traffic), fused gate+pointwise, fp32 cell state,
// double-buffered h state.

typedef __bf16 bf16_t;
typedef __attribute__((ext_vector_type(16))) __bf16 v16bf;
typedef __attribute__((ext_vector_type(8)))  __bf16 v8bf;
typedef __attribute__((ext_vector_type(8)))  float   v8f;

#define V_DIM 256
#define H_DIM 1024
#define B_DIM 64
#define S_DIM 128
#define T_DIM 127
#define G4H   4096   // 4*H

static __device__ inline v16bf cat8(v8bf lo, v8bf hi) {
  v16bf r;
#pragma unroll
  for (int i = 0; i < 8; ++i) { r[i] = lo[i]; r[i + 8] = hi[i]; }
  return r;
}
static __device__ inline float sigmoid_(float x) { return 1.f / (1.f + __expf(-x)); }
static __device__ inline float tanh_(float x)    { return 2.f / (1.f + __expf(-2.f * x)) - 1.f; }

// Load one K=32 step worth of tiles: B tile (16 cols) + 4 A tiles (M=64).
// Per-lane tile layout (ISA 16-bit A 16x32): lane<16 -> M=lane, K chunks
// k+0..7 & k+16..23; lane>=16 -> M=lane-16, K chunks k+8..15 & k+24..31.
// B tile mirrored (lane -> output column).
static __device__ inline void load_tiles(const bf16_t* __restrict__ Ab, size_t astr,
                                         const bf16_t* __restrict__ Brow, int k,
                                         v16bf& b, v16bf a[4]) {
  b = cat8(*(const v8bf*)(Brow + k), *(const v8bf*)(Brow + k + 16));
#pragma unroll
  for (int m = 0; m < 4; ++m) {
    const bf16_t* Ar = Ab + (size_t)(m * 16) * astr;
    a[m] = cat8(*(const v8bf*)(Ar + k), *(const v8bf*)(Ar + k + 16));
  }
}
static __device__ inline void wmma4(const v16bf a[4], v16bf b, v8f acc[4]) {
#pragma unroll
  for (int m = 0; m < 4; ++m)
    acc[m] = __builtin_amdgcn_wmma_f32_16x16x32_bf16(false, a[m], false, b, (short)0,
                                                     acc[m], false, false);
}

// Ping-pong double-buffered GEMM strip: acc[4] (M=64) += A(64xK) * B(Kx16).
// (KE-KS) is a multiple of 64 for every instantiation in this file, so the
// steady-state loop advances 64 K per trip with no buffer-rotation copies.
template <int KS, int KE>
static __device__ inline void gemm_strip(const bf16_t* __restrict__ Ab, size_t astr,
                                         const bf16_t* __restrict__ Brow, int koff,
                                         v8f acc[4]) {
  if (KE <= KS) return;
  v16bf b0, b1;
  v16bf a0[4], a1[4];
  load_tiles(Ab, astr, Brow, KS + koff, b0, a0);
  int kb = KS;
#pragma unroll 1
  for (; kb + 64 < KE; kb += 64) {
    __builtin_prefetch(Brow + kb + koff + 544, 0, 1);      // global_prefetch_b8
    load_tiles(Ab, astr, Brow, kb + 32 + koff, b1, a1);    // fill buf1
    wmma4(a0, b0, acc);                                    // consume buf0
    load_tiles(Ab, astr, Brow, kb + 64 + koff, b0, a0);    // refill buf0
    wmma4(a1, b1, acc);                                    // consume buf1
  }
  load_tiles(Ab, astr, Brow, kb + 32 + koff, b1, a1);      // final 64 K
  wmma4(a0, b0, acc);
  wmma4(a1, b1, acc);
}

// ---------------- prep kernels ----------------
__global__ void cvt_bf16_k(const float* __restrict__ s, bf16_t* __restrict__ d, int n) {
  for (int i = blockIdx.x * blockDim.x + threadIdx.x; i < n; i += gridDim.x * blockDim.x)
    d[i] = (bf16_t)s[i];
}

// bias layout out[((d*2+j)*2+l)*4096 + g] = bih + bhh
__global__ void bias_combine_k(const float* __restrict__ bi0, const float* __restrict__ bh0,
                               const float* __restrict__ bir, const float* __restrict__ bhr,
                               float* __restrict__ out) {
  int i = blockIdx.x * blockDim.x + threadIdx.x;   // over 2*2*4096 (d,j,g)
  if (i < 4 * G4H) {
    int pair = i / G4H, g = i % G4H;               // pair = d*2+j
    out[(pair * 2 + 0) * G4H + g] = bi0[i] + bh0[i];
    out[(pair * 2 + 1) * G4H + g] = bir[i] + bhr[i];
  }
}

__global__ void zero_state_k(bf16_t* __restrict__ h0, float* __restrict__ c, int n) {
  for (int i = blockIdx.x * blockDim.x + threadIdx.x; i < n; i += gridDim.x * blockDim.x) {
    h0[i] = (bf16_t)0.f;
    c[i]  = 0.f;
  }
}

// ---------------- fused LSTM cell ----------------
// Block = 256 threads = 8 waves. Wave w: gate g=w&3, K-half kh=w>>2.
// Each gate's two waves split the combined K (KIN input + H recurrent) at
// HB=(KIN+H)/2, so every weight element is read exactly once per block while
// doubling resident waves. Partial gate tiles reduced through LDS.
template <int KIN>
__global__ void __launch_bounds__(256)
lstm_cell_k(const bf16_t* __restrict__ A, int a_stride, int a_off0, int a_off1,
            const bf16_t* __restrict__ Wih, int wih_dstride,
            const bf16_t* __restrict__ Whh, int whh_dstride,
            const float*  __restrict__ bias, int bias_dstride,
            const bf16_t* __restrict__ hprev, int h_dstride,
            bf16_t* __restrict__ hout,
            float*  __restrict__ cstate) {
  const int w     = threadIdx.x >> 5;     // wave id 0..7
  const int g     = w & 3;                // gate
  const int kh    = w >> 2;               // K half
  const int lane  = threadIdx.x & 31;
  const int ln    = lane & 15;
  const int koff  = (lane >> 4) * 8;
  const int nbase = blockIdx.x << 4;      // H=1024 -> 64 column tiles
  const int d     = blockIdx.y;           // direction
  constexpr int HB  = (KIN + H_DIM) / 2;  // combined-K split point
  constexpr int WSP = HB - KIN;           // Whh-internal split (>=0, mult of 64)

  const bf16_t* Abase = A + (d ? a_off1 : a_off0) + (size_t)ln * a_stride;
  const bf16_t* Wrow  = Wih + (size_t)d * wih_dstride + (size_t)(g * H_DIM + nbase + ln) * KIN;
  const bf16_t* Hbase = hprev + (size_t)d * h_dstride + (size_t)ln * H_DIM;
  const bf16_t* Urow  = Whh + (size_t)d * whh_dstride + (size_t)(g * H_DIM + nbase + ln) * H_DIM;

  v8f acc[4] = {};
  if (kh == 0) {
    gemm_strip<0, KIN>(Abase, (size_t)a_stride, Wrow, koff, acc);  // v @ Wih^T
    gemm_strip<0, WSP>(Hbase, H_DIM, Urow, koff, acc);             // h @ Whh^T (front)
  } else {
    gemm_strip<WSP, H_DIM>(Hbase, H_DIM, Urow, koff, acc);         // h @ Whh^T (back)
  }

  // partial gate strips -> LDS, reduce K-halves, pointwise LSTM update
  __shared__ float gl[8][64][16];          // 32 KB
#pragma unroll
  for (int m = 0; m < 4; ++m)
#pragma unroll
    for (int r = 0; r < 8; ++r)            // C layout: lane<16 M=r, lane>=16 M=r+8
      gl[w][m * 16 + r + koff][ln] = acc[m][r];
  __syncthreads();

  const float* bb = bias + (size_t)d * bias_dstride + nbase;
  for (int e = threadIdx.x; e < B_DIM * 16; e += 256) {
    const int m = e >> 4, n = e & 15;      // m = batch row 0..63
    const float gi = gl[0][m][n] + gl[4][m][n] + bb[0 * H_DIM + n];
    const float gf = gl[1][m][n] + gl[5][m][n] + bb[1 * H_DIM + n];
    const float gg = gl[2][m][n] + gl[6][m][n] + bb[2 * H_DIM + n];
    const float go = gl[3][m][n] + gl[7][m][n] + bb[3 * H_DIM + n];
    const size_t ci = (size_t)d * h_dstride + (size_t)m * H_DIM + (nbase + n);
    const float cn = sigmoid_(gf) * cstate[ci] + sigmoid_(gi) * tanh_(gg);
    cstate[ci] = cn;
    hout[ci]   = (bf16_t)(sigmoid_(go) * tanh_(cn));
  }
}

// ---------------- FC projection H -> V ----------------
// Block = 8 waves: wave w owns V column tile nt=w&3, K half kh=w>>2.
__global__ void __launch_bounds__(256)
fc_k(const bf16_t* __restrict__ Hin, int h_dstride,
     const bf16_t* __restrict__ Wfc, int w_dstride,
     const float*  __restrict__ bfc, int b_dstride,
     bf16_t* __restrict__ vec_out,   // depth 0: bf16 feed to next depth
     float*  __restrict__ out_f32,   // depth 1: final fp32 output
     int t) {
  const int w     = threadIdx.x >> 5;
  const int nt    = w & 3;
  const int kh    = w >> 2;
  const int lane  = threadIdx.x & 31;
  const int ln    = lane & 15;
  const int koff  = (lane >> 4) * 8;
  const int nbase = (blockIdx.x * 4 + nt) << 4;      // V=256 -> 16 column tiles
  const int d     = blockIdx.y;

  const bf16_t* Hbase = Hin + (size_t)d * h_dstride + (size_t)ln * H_DIM;
  const bf16_t* Wrow  = Wfc + (size_t)d * w_dstride + (size_t)(nbase + ln) * H_DIM;
  v8f acc[4] = {};
  if (kh == 0) gemm_strip<0, H_DIM / 2>(Hbase, H_DIM, Wrow, koff, acc);
  else         gemm_strip<H_DIM / 2, H_DIM>(Hbase, H_DIM, Wrow, koff, acc);

  __shared__ float gl[8][64][16];          // 32 KB
#pragma unroll
  for (int m = 0; m < 4; ++m)
#pragma unroll
    for (int r = 0; r < 8; ++r)
      gl[w][m * 16 + r + koff][ln] = acc[m][r];
  __syncthreads();

  for (int e = threadIdx.x; e < 4 * B_DIM * 16; e += 256) {
    const int n2  = e >> 10;               // which of the 4 column tiles
    const int rem = e & 1023;
    const int m = rem >> 4, n = rem & 15;
    const int col = (blockIdx.x * 4 + n2) * 16 + n;
    const float val = gl[n2][m][n] + gl[n2 + 4][m][n] + bfc[(size_t)d * b_dstride + col];
    if (out_f32) {
      out_f32[(size_t)m * (2 * T_DIM * V_DIM) + (size_t)(d * T_DIM + t) * V_DIM + col] = val;
    } else {
      vec_out[(size_t)d * (B_DIM * V_DIM) + (size_t)m * V_DIM + col] = (bf16_t)val;
    }
  }
}

// ---------------- host ----------------
extern "C" void kernel_launch(void* const* d_in, const int* in_sizes, int n_in,
                              void* d_out, int out_size, void* d_ws, size_t ws_size,
                              hipStream_t stream) {
  (void)in_sizes; (void)n_in; (void)out_size; (void)ws_size;
  const float* x    = (const float*)d_in[0];
  const float* Wih0 = (const float*)d_in[1];
  const float* Whh0 = (const float*)d_in[2];
  const float* bih0 = (const float*)d_in[3];
  const float* bhh0 = (const float*)d_in[4];
  const float* Wihr = (const float*)d_in[5];
  const float* Whhr = (const float*)d_in[6];
  const float* bihr = (const float*)d_in[7];
  const float* bhhr = (const float*)d_in[8];
  const float* fcW  = (const float*)d_in[9];
  const float* fcb  = (const float*)d_in[10];
  float* out = (float*)d_out;

  // workspace carve-up (~120 MB): bf16 weights + states
  uint8_t* ws = (uint8_t*)d_ws;
  size_t off = 0;
  auto alloc = [&](size_t bytes) -> void* {
    void* p = ws + off;
    off = (off + bytes + 255) & ~(size_t)255;
    return p;
  };
  const int nX    = B_DIM * S_DIM * V_DIM;       // 2,097,152
  const int nWih0 = 2 * 2 * G4H * V_DIM;         // 4,194,304
  const int nWhh  = 2 * 2 * G4H * H_DIM;         // 16,777,216
  const int nFcW  = 2 * 2 * V_DIM * H_DIM;       // 1,048,576
  const int nState = 2 * 2 * 2 * B_DIM * H_DIM;  // [d][j][l][B][H] = 524,288

  bf16_t* xbf   = (bf16_t*)alloc((size_t)nX * 2);
  bf16_t* Wih0b = (bf16_t*)alloc((size_t)nWih0 * 2);
  bf16_t* Whh0b = (bf16_t*)alloc((size_t)nWhh * 2);
  bf16_t* Wih1b = (bf16_t*)alloc((size_t)nWhh * 2);
  bf16_t* Whh1b = (bf16_t*)alloc((size_t)nWhh * 2);
  bf16_t* fcWb  = (bf16_t*)alloc((size_t)nFcW * 2);
  float*  biasc = (float*)alloc((size_t)8 * G4H * 4);     // [d][j][l][4H]
  bf16_t* hbank = (bf16_t*)alloc((size_t)2 * nState * 2); // ping-pong h (bf16)
  float*  cst   = (float*)alloc((size_t)nState * 4);      // cell state (f32)
  bf16_t* vecb  = (bf16_t*)alloc((size_t)2 * B_DIM * V_DIM * 2); // inter-depth vec

  auto cvt = [&](const float* s, bf16_t* d, int n) {
    int blocks = (n + 255) / 256;
    if (blocks > 4096) blocks = 4096;
    cvt_bf16_k<<<blocks, 256, 0, stream>>>(s, d, n);
  };
  cvt(x, xbf, nX);
  cvt(Wih0, Wih0b, nWih0);
  cvt(Whh0, Whh0b, nWhh);
  cvt(Wihr, Wih1b, nWhh);
  cvt(Whhr, Whh1b, nWhh);
  cvt(fcW, fcWb, nFcW);
  bias_combine_k<<<(4 * G4H + 255) / 256, 256, 0, stream>>>(bih0, bhh0, bihr, bhhr, biasc);
  zero_state_k<<<2048, 256, 0, stream>>>(hbank, cst, nState);

  const int hBlk = B_DIM * H_DIM;     // 65536 per (d,j,l) block
  const int hDir = 4 * hBlk;          // dir stride within a bank
  const dim3 gridL(64, 2), gridF(4, 2), blk(256);

  for (int t = 0; t < T_DIM; ++t) {
    bf16_t* hR = hbank + (size_t)(t & 1) * nState;        // read bank
    bf16_t* hW = hbank + (size_t)((t & 1) ^ 1) * nState;  // write bank
    for (int j = 0; j < 2; ++j) {
      // ---- layer 0 (input dim V) ----
      const bf16_t* A0; int aStr, aOff0, aOff1;
      if (j == 0) { A0 = xbf;  aStr = S_DIM * V_DIM; aOff0 = t * V_DIM; aOff1 = (S_DIM - 1 - t) * V_DIM; }
      else        { A0 = vecb; aStr = V_DIM;         aOff0 = 0;         aOff1 = B_DIM * V_DIM; }
      lstm_cell_k<V_DIM><<<gridL, blk, 0, stream>>>(
          A0, aStr, aOff0, aOff1,
          Wih0b + (size_t)j * G4H * V_DIM, 2 * G4H * V_DIM,
          Whh0b + (size_t)j * G4H * H_DIM, 2 * G4H * H_DIM,
          biasc + (j * 2 + 0) * G4H, 4 * G4H,
          hR + (j * 2 + 0) * hBlk, hDir,
          hW + (j * 2 + 0) * hBlk,
          cst + (j * 2 + 0) * hBlk);
      // ---- layer 1 (input dim H) ----
      lstm_cell_k<H_DIM><<<gridL, blk, 0, stream>>>(
          hW + (j * 2 + 0) * hBlk, H_DIM, 0, hDir,
          Wih1b + (size_t)j * G4H * H_DIM, 2 * G4H * H_DIM,
          Whh1b + (size_t)j * G4H * H_DIM, 2 * G4H * H_DIM,
          biasc + (j * 2 + 1) * G4H, 4 * G4H,
          hR + (j * 2 + 1) * hBlk, hDir,
          hW + (j * 2 + 1) * hBlk,
          cst + (j * 2 + 1) * hBlk);
      // ---- FC projection ----
      fc_k<<<gridF, blk, 0, stream>>>(
          hW + (j * 2 + 1) * hBlk, hDir,
          fcWb + (size_t)j * V_DIM * H_DIM, 2 * V_DIM * H_DIM,
          fcb + j * V_DIM, 2 * V_DIM,
          (j == 0) ? vecb : (bf16_t*)nullptr,
          (j == 0) ? (float*)nullptr : out, t);
    }
  }
}
